// RegularGrid_34256659153446
// MI455X (gfx1250) — matrix operand order repaired and checked by reading, not compiled, not run on hardware
//
#include <hip/hip_runtime.h>
#include <math.h>

// ---------------------------------------------------------------------------
// Plenoxels-style ray marcher for MI455X (gfx1250, wave32).
//
// Phase 1 (repack, every launch): grid (28, 128^3) f32 channel-major ->
//   (128^3, 32) f16 voxel-major in d_ws. Packed channel order is
//   [SH0..SH14, sigma, SH15..SH26, 0,0,0,0] so sigma lands at K=15 (the low
//   lane's last B element) -> no cross-half shuffle in the march loop.
//
// Phase 2 (march): 1 wave = 1 ray; lane pair (n, n+16) shares sample n:
//   low lane = packed ch 0..15 (bytes 0..31), high lane = ch 16..31.
//   Each corner gather = 2x global_load_b128 per lane; trilerp accumulates
//   with v_pk_fma_f16 directly into the WMMA B-operand layout.
//   SH projection = V_WMMA_F32_16X16X32_F16 (A row = color, K = packed ch).
//   Transmittance scan & reduction = v_permlane16_b32 (no LDS round-trips).
// ---------------------------------------------------------------------------

typedef __attribute__((ext_vector_type(16))) _Float16 v16h;
typedef __attribute__((ext_vector_type(8)))  _Float16 v8h;
typedef __attribute__((ext_vector_type(8)))  float    v8f;

#define RADIUS_F   1.3f
#define STEP_F     (RADIUS_F / 128.0f)     // 2*R/RES/2
#define N_SAMP     443                     // N_INTRS-1, N_INTRS = ceil(sqrt(3)*256)
#define N_ITER     28                      // ceil(443/16)
#define N_RAYS     4096
#define VOX_SHIFT  21                      // 128^3 = 2^21 voxels per channel
#define N_VOX      (1 << VOX_SHIFT)
#define PACK_CH    32                      // padded channel count in repacked grid
#define PACK_BYTES ((size_t)N_VOX * PACK_CH * 2)

union HV  { v16h v; v8h p[2]; _Float16 h[16]; };
union FV  { v8f  v; float    f[8];  };

__device__ __forceinline__ float sigmoidf_fast(float x) {
    return __builtin_amdgcn_rcpf(1.0f + __expf(-x));   // v_rcp_f32 path
}

// ---- v_permlane16_b32 helpers (16-lane permutes, pure VALU) ---------------
#if __has_builtin(__builtin_amdgcn_permlane16)
__device__ __forceinline__ float permlane16_f(float x, unsigned s0, unsigned s1) {
    int xi = __builtin_bit_cast(int, x);
    int r  = __builtin_amdgcn_permlane16(xi, xi, s0, s1, false, false);
    return __builtin_bit_cast(float, r);
}
#define PL_UP1(x)   permlane16_f((x), 0x65432100u, 0xEDCBA987u)
#define PL_UP2(x)   permlane16_f((x), 0x54321000u, 0xDCBA9876u)
#define PL_UP4(x)   permlane16_f((x), 0x32100000u, 0xBA987654u)
#define PL_UP8(x)   permlane16_f((x), 0x00000000u, 0x76543210u)
#define PL_DOWN8(x) permlane16_f((x), 0xFEDCBA98u, 0xFEDCBA98u)
#define PL_DOWN4(x) permlane16_f((x), 0xBA987654u, 0xFEDCFEDCu)
#define PL_DOWN2(x) permlane16_f((x), 0x98765432u, 0xFFFEDCBAu)
#define PL_DOWN1(x) permlane16_f((x), 0x87654321u, 0xFFEDCBA9u)
#define PL_BC15(x)  permlane16_f((x), 0xFFFFFFFFu, 0xFFFFFFFFu)
#else
#define PL_UP1(x)   __shfl_up((x), 1, 16)
#define PL_UP2(x)   __shfl_up((x), 2, 16)
#define PL_UP4(x)   __shfl_up((x), 4, 16)
#define PL_UP8(x)   __shfl_up((x), 8, 16)
#define PL_DOWN8(x) __shfl_down((x), 8, 16)
#define PL_DOWN4(x) __shfl_down((x), 4, 16)
#define PL_DOWN2(x) __shfl_down((x), 2, 16)
#define PL_DOWN1(x) __shfl_down((x), 1, 16)
#define PL_BC15(x)  __shfl((x), 15, 16)
#endif

// Packed-channel -> original-channel map: K<15 -> K, K==15 -> sigma(27),
// 16<=K<=27 -> K-1, K>=28 -> pad.
__device__ __host__ __forceinline__ int pack_src_channel(int k) {
    return (k < 15) ? k : (k == 15 ? 27 : (k <= 27 ? k - 1 : -1));
}

// ---------------------------------------------------------------------------
// Phase 1: repack (c,z,y,x) f32 -> (z,y,x,c32) f16 with channel reorder.
// ---------------------------------------------------------------------------
__global__ __launch_bounds__(256)
void repack_kernel(const float* __restrict__ grid, _Float16* __restrict__ pack)
{
    const int v = blockIdx.x * 256 + threadIdx.x;     // voxel id 0..2^21-1
    _Float16 tmp[PACK_CH];
    #pragma unroll
    for (int k = 0; k < PACK_CH; ++k) {
        const int src = pack_src_channel(k);
        tmp[k] = (src >= 0) ? (_Float16)grid[((size_t)src << VOX_SHIFT) + v]
                            : (_Float16)0.0f;
    }
    v8h* dst = (v8h*)(pack + (size_t)v * PACK_CH);
    #pragma unroll
    for (int q = 0; q < 4; ++q) {
        v8h w;
        #pragma unroll
        for (int e = 0; e < 8; ++e) w[e] = tmp[q * 8 + e];
        dst[q] = w;
    }
}

// ---------------------------------------------------------------------------
// A matrix (16x32 f16) in contiguous K order (same as B); remap: uses packed channel order if
// `packed`, else identity (channels 0..26 SH, 27 sigma).
//  lanes 0-15 : M=lane,    VGPR0-7 -> K=0..15
//  lanes16-31 : M=lane-16, VGPR0-7 -> K=16..31
// ---------------------------------------------------------------------------
__device__ __forceinline__ void build_A(HV& A, const float* shm, int sp, bool lo,
                                        bool packed)
{
    #pragma unroll
    for (int e = 0; e < 16; ++e) A.h[e] = (_Float16)0.0f;
    if (sp < 3) {
        #pragma unroll
        for (int e = 0; e < 16; ++e) {
            const int K  = lo ? e : e + 16;
            const int ch = packed ? pack_src_channel(K) : K;
            if (ch >= 0 && ch < 27) {
                const int j = ch - 9 * sp;
                if (j >= 0 && j < 9) A.h[e] = (_Float16)shm[j];
            }
        }
    }
}

__device__ __forceinline__ void sh_basis(float* shm, float dx, float dy, float dz)
{
    shm[0] =  0.28209479177387814f;
    shm[1] = -0.4886025119029199f  * dy;
    shm[2] =  0.4886025119029199f  * dz;
    shm[3] = -0.4886025119029199f  * dx;
    shm[4] =  1.0925484305920792f  * dx * dy;
    shm[5] = -1.0925484305920792f  * dy * dz;
    shm[6] =  0.31539156525252005f * (2.0f*dz*dz - dx*dx - dy*dy);
    shm[7] = -1.0925484305920792f  * dx * dz;
    shm[8] =  0.5462742152960396f  * (dx*dx - dy*dy);
}

// ---------------------------------------------------------------------------
// Phase 2 (fast path): march over the repacked f16 voxel-major grid.
// ---------------------------------------------------------------------------
__global__ __launch_bounds__(256)
void raymarch_packed_kernel(const float* __restrict__ rays_o,
                            const float* __restrict__ rays_d,
                            const _Float16* __restrict__ pack,
                            float* __restrict__ out)
{
    const int wave = threadIdx.x >> 5;
    const int lane = threadIdx.x & 31;
    const int ray  = blockIdx.x * 8 + wave;
    const int sp   = lane & 15;
    const bool lo  = (lane < 16);

    const float ox = rays_o[ray*3+0], oy = rays_o[ray*3+1], oz = rays_o[ray*3+2];
    const float dx = rays_d[ray*3+0], dy = rays_d[ray*3+1], dz = rays_d[ray*3+2];

    const float sx = fminf((RADIUS_F - ox)/dx, (-RADIUS_F - ox)/dx);
    const float sy = fminf((RADIUS_F - oy)/dy, (-RADIUS_F - oy)/dy);
    const float sz = fminf((RADIUS_F - oz)/dz, (-RADIUS_F - oz)/dz);
    const float start = fmaxf(sx, fmaxf(sy, sz));
    const float dist  = STEP_F * sqrtf(dx*dx + dy*dy + dz*dz);

    float shm[9];
    sh_basis(shm, dx, dy, dz);
    HV A;
    build_A(A, shm, sp, lo, /*packed=*/true);

    const v8f czero = {0.f,0.f,0.f,0.f,0.f,0.f,0.f,0.f};
    const v8h* __restrict__ P = (const v8h*)pack;     // voxel = 4 consecutive v8h
    const int laneHalf = lo ? 0 : 2;

    float T    = 1.0f;
    float accR = 0.f, accG = 0.f, accB = 0.f, accA = 0.f;

    for (int it = 0; it < N_ITER; ++it) {
        const int   s  = it * 16 + sp;
        const float t  = start + (float)s * STEP_F;
        const float px = ox + t*dx, py = oy + t*dy, pz = oz + t*dz;

        const bool inBox = (s < N_SAMP)
            && (px > -RADIUS_F) && (px < RADIUS_F)
            && (py > -RADIUS_F) && (py < RADIUS_F)
            && (pz > -RADIUS_F) && (pz < RADIUS_F);

        const float cx = fminf(fmaxf((px/RADIUS_F + 1.f) * 63.5f, 0.f), 127.f);
        const float cy = fminf(fmaxf((py/RADIUS_F + 1.f) * 63.5f, 0.f), 127.f);
        const float cz = fminf(fmaxf((pz/RADIUS_F + 1.f) * 63.5f, 0.f), 127.f);
        const int ix = (int)cx, iy = (int)cy, iz = (int)cz;
        const float fx = cx - (float)ix, fy = cy - (float)iy, fz = cz - (float)iz;

        const int vbase = (((iz << 14) + (iy << 7) + ix) << 2) + laneHalf;
        const int dxv = (ix < 127) ? 4     : 0;       // +1 voxel in x (v8h units)
        const int dyv = (iy < 127) ? 512   : 0;       // +1 voxel in y
        const int dzv = (iz < 127) ? 65536 : 0;       // +1 voxel in z

        const float wx0 = 1.f - fx, wy0 = 1.f - fy, wz0 = 1.f - fz;
        const _Float16 w000 = (_Float16)(wz0*wy0*wx0), w001 = (_Float16)(wz0*wy0*fx);
        const _Float16 w010 = (_Float16)(wz0*fy*wx0),  w011 = (_Float16)(wz0*fy*fx);
        const _Float16 w100 = (_Float16)(fz*wy0*wx0),  w101 = (_Float16)(fz*wy0*fx);
        const _Float16 w110 = (_Float16)(fz*fy*wx0),   w111 = (_Float16)(fz*fy*fx);

        HV B;
        B.p[0] = (v8h)(_Float16)0.0f;
        B.p[1] = (v8h)(_Float16)0.0f;
        if (inBox) {
            const int o[8] = { vbase,             vbase + dxv,
                               vbase + dyv,       vbase + dyv + dxv,
                               vbase + dzv,       vbase + dzv + dxv,
                               vbase + dzv + dyv, vbase + dzv + dyv + dxv };
            v8h c0[8], c1[8];
            #pragma unroll
            for (int j = 0; j < 8; ++j) { c0[j] = P[o[j]]; c1[j] = P[o[j] + 1]; }

            v8h a0, a1;
            a0  = c0[0] * w000;  a1  = c1[0] * w000;
            a0 += c0[1] * w001;  a1 += c1[1] * w001;
            a0 += c0[2] * w010;  a1 += c1[2] * w010;
            a0 += c0[3] * w011;  a1 += c1[3] * w011;
            a0 += c0[4] * w100;  a1 += c1[4] * w100;
            a0 += c0[5] * w101;  a1 += c1[5] * w101;
            a0 += c0[6] * w110;  a1 += c1[6] * w110;
            a0 += c0[7] * w111;  a1 += c1[7] * w111;
            B.p[0] = a0;
            B.p[1] = a1;
        }

        // EXEC fully reconverged: WMMA needs all 32 lanes active.
        FV D;
        D.v = __builtin_amdgcn_wmma_f32_16x16x32_f16(
                  false, A.v, false, B.v, (short)0, czero, false, false);

        // sigma sits at K=15 of THIS lane's B half (low lanes; hi lanes get pad=0)
        float sigma = fmaxf((float)B.h[15], 0.f);
        const float alpha = 1.f - __expf(-sigma * dist);

        // exclusive product scan of g over 16 samples, all v_permlane16.
        const float g = 1.f - alpha + 1e-10f;
        float h = PL_UP1(g);
        h = (sp == 0) ? 1.f : h;
        float p = h, y;
        y = PL_UP1(p); p = (sp >= 1) ? p * y : p;
        y = PL_UP2(p); p = (sp >= 2) ? p * y : p;
        y = PL_UP4(p); p = (sp >= 4) ? p * y : p;
        y = PL_UP8(p); p = (sp >= 8) ? p * y : p;

        const float weight = alpha * T * p;
        T *= PL_BC15(p * g);

        if (lo) {
            accR += weight * sigmoidf_fast(D.f[0]);
            accG += weight * sigmoidf_fast(D.f[1]);
            accB += weight * sigmoidf_fast(D.f[2]);
            accA += weight;
        }
    }

    // reduce partials across the 16 sample lanes (permlane16 tree).
    accR += PL_DOWN8(accR); accG += PL_DOWN8(accG); accB += PL_DOWN8(accB); accA += PL_DOWN8(accA);
    accR += PL_DOWN4(accR); accG += PL_DOWN4(accG); accB += PL_DOWN4(accB); accA += PL_DOWN4(accA);
    accR += PL_DOWN2(accR); accG += PL_DOWN2(accG); accB += PL_DOWN2(accB); accA += PL_DOWN2(accA);
    accR += PL_DOWN1(accR); accG += PL_DOWN1(accG); accB += PL_DOWN1(accB); accA += PL_DOWN1(accA);

    if (lane == 0) {
        const float bg = 1.f - accA;
        out[ray*3+0] = accR + bg;
        out[ray*3+1] = accG + bg;
        out[ray*3+2] = accB + bg;
    }
}

// ---------------------------------------------------------------------------
// Fallback: direct gather from the f32 channel-major grid (no workspace).
// ---------------------------------------------------------------------------
__global__ __launch_bounds__(256)
void raymarch_direct_kernel(const float* __restrict__ rays_o,
                            const float* __restrict__ rays_d,
                            const float* __restrict__ grid,
                            float* __restrict__ out)
{
    const int wave = threadIdx.x >> 5;
    const int lane = threadIdx.x & 31;
    const int ray  = blockIdx.x * 8 + wave;
    const int sp   = lane & 15;
    const bool lo  = (lane < 16);

    const float ox = rays_o[ray*3+0], oy = rays_o[ray*3+1], oz = rays_o[ray*3+2];
    const float dx = rays_d[ray*3+0], dy = rays_d[ray*3+1], dz = rays_d[ray*3+2];

    const float sx = fminf((RADIUS_F - ox)/dx, (-RADIUS_F - ox)/dx);
    const float sy = fminf((RADIUS_F - oy)/dy, (-RADIUS_F - oy)/dy);
    const float sz = fminf((RADIUS_F - oz)/dz, (-RADIUS_F - oz)/dz);
    const float start = fmaxf(sx, fmaxf(sy, sz));
    const float dist  = STEP_F * sqrtf(dx*dx + dy*dy + dz*dz);

    float shm[9];
    sh_basis(shm, dx, dy, dz);
    HV A;
    build_A(A, shm, sp, lo, /*packed=*/false);

    const v8f czero = {0.f,0.f,0.f,0.f,0.f,0.f,0.f,0.f};
    float T    = 1.0f;
    float accR = 0.f, accG = 0.f, accB = 0.f, accA = 0.f;
    const int chBase  = lo ? 0  : 16;
    const int chCount = lo ? 16 : 12;

    for (int it = 0; it < N_ITER; ++it) {
        const int   s  = it * 16 + sp;
        const float t  = start + (float)s * STEP_F;
        const float px = ox + t*dx, py = oy + t*dy, pz = oz + t*dz;

        const bool inBox = (s < N_SAMP)
            && (px > -RADIUS_F) && (px < RADIUS_F)
            && (py > -RADIUS_F) && (py < RADIUS_F)
            && (pz > -RADIUS_F) && (pz < RADIUS_F);

        const float cx = fminf(fmaxf((px/RADIUS_F + 1.f) * 63.5f, 0.f), 127.f);
        const float cy = fminf(fmaxf((py/RADIUS_F + 1.f) * 63.5f, 0.f), 127.f);
        const float cz = fminf(fmaxf((pz/RADIUS_F + 1.f) * 63.5f, 0.f), 127.f);
        const int ix = (int)cx, iy = (int)cy, iz = (int)cz;
        const float fx = cx - (float)ix, fy = cy - (float)iy, fz = cz - (float)iz;

        const int o000 = (iz << 14) + (iy << 7) + ix;
        const int ddx  = (ix < 127) ? 1     : 0;
        const int ddy  = (iy < 127) ? 128   : 0;
        const int ddz  = (iz < 127) ? 16384 : 0;

        const float wx0 = 1.f - fx, wy0 = 1.f - fy, wz0 = 1.f - fz;
        const float w000 = wz0*wy0*wx0, w001 = wz0*wy0*fx;
        const float w010 = wz0*fy*wx0,  w011 = wz0*fy*fx;
        const float w100 = fz*wy0*wx0,  w101 = fz*wy0*fx;
        const float w110 = fz*fy*wx0,   w111 = fz*fy*fx;

        HV B;
        B.p[0] = (v8h)(_Float16)0.0f;
        B.p[1] = (v8h)(_Float16)0.0f;
        float sigmaRaw = 0.f;
        if (inBox) {
            for (int e = 0; e < chCount; ++e) {
                const float* gp = grid + ((size_t)(chBase + e) << VOX_SHIFT) + o000;
                const float v =
                      w000*gp[0]       + w001*gp[ddx]
                    + w010*gp[ddy]     + w011*gp[ddy+ddx]
                    + w100*gp[ddz]     + w101*gp[ddz+ddx]
                    + w110*gp[ddz+ddy] + w111*gp[ddz+ddy+ddx];
                B.h[e] = (_Float16)v;
                if (!lo && e == 11) sigmaRaw = v;   // original ch 27 = sigma
            }
        }

        FV D;
        D.v = __builtin_amdgcn_wmma_f32_16x16x32_f16(
                  false, A.v, false, B.v, (short)0, czero, false, false);

        float sigma = __shfl_down(sigmaRaw, 16, 32);
        sigma = fmaxf(sigma, 0.f);
        const float alpha = 1.f - __expf(-sigma * dist);

        const float g = 1.f - alpha + 1e-10f;
        float h = PL_UP1(g);
        h = (sp == 0) ? 1.f : h;
        float p = h, y;
        y = PL_UP1(p); p = (sp >= 1) ? p * y : p;
        y = PL_UP2(p); p = (sp >= 2) ? p * y : p;
        y = PL_UP4(p); p = (sp >= 4) ? p * y : p;
        y = PL_UP8(p); p = (sp >= 8) ? p * y : p;

        const float weight = alpha * T * p;
        T *= PL_BC15(p * g);

        if (lo) {
            accR += weight * sigmoidf_fast(D.f[0]);
            accG += weight * sigmoidf_fast(D.f[1]);
            accB += weight * sigmoidf_fast(D.f[2]);
            accA += weight;
        }
    }

    accR += PL_DOWN8(accR); accG += PL_DOWN8(accG); accB += PL_DOWN8(accB); accA += PL_DOWN8(accA);
    accR += PL_DOWN4(accR); accG += PL_DOWN4(accG); accB += PL_DOWN4(accB); accA += PL_DOWN4(accA);
    accR += PL_DOWN2(accR); accG += PL_DOWN2(accG); accB += PL_DOWN2(accB); accA += PL_DOWN2(accA);
    accR += PL_DOWN1(accR); accG += PL_DOWN1(accG); accB += PL_DOWN1(accB); accA += PL_DOWN1(accA);

    if (lane == 0) {
        const float bg = 1.f - accA;
        out[ray*3+0] = accR + bg;
        out[ray*3+1] = accG + bg;
        out[ray*3+2] = accB + bg;
    }
}

extern "C" void kernel_launch(void* const* d_in, const int* in_sizes, int n_in,
                              void* d_out, int out_size, void* d_ws, size_t ws_size,
                              hipStream_t stream) {
    (void)in_sizes; (void)n_in; (void)out_size;
    const float* rays_o = (const float*)d_in[0];   // (4096,3) f32
    const float* rays_d = (const float*)d_in[1];   // (4096,3) f32
    const float* data   = (const float*)d_in[2];   // (1,28,128,128,128) f32
    float* out = (float*)d_out;                    // (4096,3) f32

    dim3 block(256);
    dim3 mgrid(N_RAYS / 8);                        // 8 rays (waves) per block

    if (d_ws != nullptr && ws_size >= PACK_BYTES) {
        _Float16* pack = (_Float16*)d_ws;
        repack_kernel<<<dim3(N_VOX / 256), block, 0, stream>>>(data, pack);
        raymarch_packed_kernel<<<mgrid, block, 0, stream>>>(rays_o, rays_d, pack, out);
    } else {
        raymarch_direct_kernel<<<mgrid, block, 0, stream>>>(rays_o, rays_d, data, out);
    }
}
